// TemporalMixMultiHeadAttention_3126736192156
// MI455X (gfx1250) — matrix operand order repaired and checked by reading, not compiled, not run on hardware
//
#include <hip/hip_runtime.h>

// ---------------------------------------------------------------------------
// TemporalMixMultiHeadAttention for MI455X (gfx1250, wave32, WMMA bf16/f32acc)
//
// Shapes: B=8, S=128 (SEED=64 + TGT=64), T=64, D=256, H=8, DK=32
// Pipeline: weights->bf16T | QKV proj GEMM | seed attn | target attn | out GEMM
// Workspace layout (bytes):
//   [0,       512K)  : Wt q/k/v/d  (bf16, transposed: Wt[k*256+n] = W[n][k])
//   [512K,  +32MiB)  : qp  bf16 [B,T,S,D]
//   [..,    +32MiB)  : kT  bf16 [B,T,H,DK,S]   (K, transposed per head)
//   [..,    +32MiB)  : vp  bf16 [B,T,S,D]
//   [..,    +32MiB)  : xo  bf16 [B,S,T,D]      (attention output)
// Total ~128.5 MiB.
// ---------------------------------------------------------------------------

typedef __attribute__((ext_vector_type(16))) __bf16 v16bf;
typedef __attribute__((ext_vector_type(8)))  float  v8f;

namespace {
constexpr int cB = 8, cS = 128, cT = 64, cD = 256, cH = 8, cDK = 32, cSEED = 64;
}

union Frag {
    v16bf          v;
    unsigned int   u32[8];
    unsigned short u16[16];
    __bf16         b16[16];
};

// Native bf16 convert (v_cvt_pk_bf16_f32 on gfx1250)
__device__ __forceinline__ unsigned short f2bf(float f) {
    union { __bf16 h; unsigned short u; } c;
    c.h = (__bf16)f;
    return c.u;
}

// A-fragment K index for 16-bit A 16x32 (lane=M, K-pairs across 8 VGPRs)
__device__ __forceinline__ int aFragK(int j, int half) {
    return ((j >= 4) ? 16 : 0) + half * 8 + (j & 3) * 2;
}

// ---------------------------------------------------------------------------
// Kernel 1: W[n][k] f32 -> Wt[k][n] bf16, for Wq,Wk,Wv,Wd
// ---------------------------------------------------------------------------
__global__ __launch_bounds__(256) void k_wt(const float* __restrict__ Wq,
                                            const float* __restrict__ Wk,
                                            const float* __restrict__ Wv,
                                            const float* __restrict__ Wd,
                                            unsigned short* __restrict__ wt) {
    int id  = blockIdx.x * 256 + threadIdx.x;   // 0 .. 4*65536-1
    int mat = id >> 16;
    int idx = id & 65535;
    int n = idx >> 8, k = idx & 255;
    const float* W = (mat == 0) ? Wq : (mat == 1) ? Wk : (mat == 2) ? Wv : Wd;
    wt[mat * 65536 + k * 256 + n] = f2bf(W[n * 256 + k]);
}

// ---------------------------------------------------------------------------
// Kernel 2: QKV projection. P[row,n] = sum_k X[row,k] * W[n,k], bf16 out.
// One wave = one 16-row M-tile x all 16 N-tiles; K=256 as two interleaved
// 4-step WMMA chains. Stores are branchless (uniform layout select) so no
// exec-mask juggling appears in the N-tile loop. K goes to a per-head
// transposed layout for the attention kernel.
// ---------------------------------------------------------------------------
__global__ __launch_bounds__(256) void k_proj(const float* __restrict__ q_in,
                                              const float* __restrict__ k_in,
                                              const float* __restrict__ v_in,
                                              const unsigned short* __restrict__ wt,
                                              unsigned short* __restrict__ qp,
                                              unsigned short* __restrict__ kT,
                                              unsigned short* __restrict__ vp) {
    const int wid = threadIdx.x >> 5, lane = threadIdx.x & 31;
    const int gw    = blockIdx.x * 8 + wid;       // 0 .. 3*4096-1
    const int mat   = gw >> 12;                   // 0..2
    const int mtile = gw & 4095;

    const float*          X  = (mat == 0) ? q_in : (mat == 1) ? k_in : v_in;
    const unsigned short* Wt = wt + mat * 65536;
    const bool isK = (mat == 1);
    unsigned short* dst = isK ? kT : ((mat == 0) ? qp : vp);

    const int half = lane >> 4, mrow = lane & 15, ncl = lane & 15;

    // Decode this lane's A row: R over ((b*T + t)*S + s)
    const int R  = mtile * 16 + mrow;
    const int b  = R >> 13;            // / (T*S)
    const int t  = (R >> 7) & 63;
    const int s  = R & 127;
    const float* xrow = X + (size_t)(((b * cS + s) * cT + t)) * cD;

    // Load all A fragments (K = 0..255), fp32 -> bf16 in-register
    Frag a[8];
#pragma unroll
    for (int ks = 0; ks < 8; ++ks) {
#pragma unroll
        for (int j = 0; j < 8; ++j) {
            const int k = ks * 32 + aFragK(j, half);
            const float2 f = *(const float2*)(xrow + k);
            a[ks].b16[2 * j]     = (__bf16)f.x;
            a[ks].b16[2 * j + 1] = (__bf16)f.y;
        }
    }

    for (int nt = 0; nt < 16; ++nt) {
        v8f c0 = {}, c1 = {};
#pragma unroll
        for (int ks = 0; ks < 4; ++ks) {
            Frag bf0, bf1;
            const int k0 = ks * 32;               // B: lane = N
            const int k1 = (ks + 4) * 32;
#pragma unroll
            for (int j = 0; j < 8; ++j) {         //    VGPR j = K pair (A-fragment order)
                const int kf = aFragK(j, half);
                bf0.u16[2 * j]     = Wt[(k0 + kf) * 256 + nt * 16 + ncl];
                bf0.u16[2 * j + 1] = Wt[(k0 + kf + 1) * 256 + nt * 16 + ncl];
                bf1.u16[2 * j]     = Wt[(k1 + kf) * 256 + nt * 16 + ncl];
                bf1.u16[2 * j + 1] = Wt[(k1 + kf + 1) * 256 + nt * 16 + ncl];
            }
            c0 = __builtin_amdgcn_wmma_f32_16x16x32_bf16(false, a[ks].v, false, bf0.v,
                                                         (short)0, c0, false, false);
            c1 = __builtin_amdgcn_wmma_f32_16x16x32_bf16(false, a[ks + 4].v, false, bf1.v,
                                                         (short)0, c1, false, false);
        }
        const int ncol = nt * 16 + ncl;
        const int h = ncol >> 5, dk = ncol & 31;  // head split (used when isK)
#pragma unroll
        for (int r = 0; r < 8; ++r) {
            const int Ro = mtile * 16 + r + half * 8;   // C: VGPR r -> M row
            const int bt = Ro >> 7, so = Ro & 127;
            const size_t idxN = (size_t)Ro * cD + ncol;
            const size_t idxT = (size_t)(bt * cH + h) * (cDK * cS) + dk * cS + so;
            dst[isK ? idxT : idxN] = f2bf(c0[r] + c1[r]);
        }
    }
}

// ---------------------------------------------------------------------------
// Kernel 3: attention. One wave per (b,t,h). TGT=0: seed self-attn (64 keys,
// key-only mask). TGT=1: target->full attn (128 keys, temporal & mix mask).
// kT/V B-fragments and temporal-mask bits are m-tile invariant and hoisted
// out of the query-tile loop. Softmax in f32 registers (shfl_xor over 16-lane
// halves), P staged bf16 in LDS for the C->A relayout, PV with dual chains.
// ---------------------------------------------------------------------------
template <int TGT>
__global__ __launch_bounds__(128) void k_attn(const unsigned short* __restrict__ qp,
                                              const unsigned short* __restrict__ kT,
                                              const unsigned short* __restrict__ vp,
                                              const int* __restrict__ tmask,
                                              const int* __restrict__ mixmask,
                                              unsigned short* __restrict__ xo) {
    __shared__ unsigned short p_lds[4][16][128];

    const int wid = threadIdx.x >> 5, lane = threadIdx.x & 31;
    const int gw = blockIdx.x * 4 + wid;            // 0..4095
    const int h = gw & 7, t = (gw >> 3) & 63, b = gw >> 9;

    constexpr int LK = TGT ? 128 : 64;
    constexpr int NT = LK / 16;
    constexpr int KK = LK / 32;

    const int half = lane >> 4, mrow = lane & 15, ncl = lane & 15;
    const size_t base = (size_t)(b * cT + t) * (size_t)(cS * cD) + h * cDK;
    const unsigned short* kTh = kT + (size_t)((b * cT + t) * cH + h) * (cDK * cS);
    const float scale = 0.17677669529663687f;       // 1/sqrt(DK)

    // ---- hoisted: kT score B-fragments (lane = key, VGPR j = d pair) -------
    Frag kb[NT];
#pragma unroll
    for (int nt = 0; nt < NT; ++nt)
#pragma unroll
        for (int j = 0; j < 8; ++j) {
            const int kf = aFragK(j, half);
            kb[nt].u16[2 * j]     = kTh[kf * cS + nt * 16 + ncl];
            kb[nt].u16[2 * j + 1] = kTh[(kf + 1) * cS + nt * 16 + ncl];
        }

    // ---- hoisted: V PV B-fragments (lane = dk, VGPR j = key pair) ----------
    Frag vbf[2 * KK];
#pragma unroll
    for (int dn = 0; dn < 2; ++dn)
#pragma unroll
        for (int kk = 0; kk < KK; ++kk)
#pragma unroll
            for (int j = 0; j < 8; ++j) {
                const int kf = aFragK(j, half);
                vbf[dn * KK + kk].u16[2 * j] =
                    vp[base + (size_t)(kk * 32 + kf) * cD + dn * 16 + ncl];
                vbf[dn * KK + kk].u16[2 * j + 1] =
                    vp[base + (size_t)(kk * 32 + kf + 1) * cD + dn * 16 + ncl];
            }

    // ---- hoisted: per-key temporal mask (this lane's key per N-tile) -------
    int tv[NT];
#pragma unroll
    for (int nt = 0; nt < NT; ++nt) {
        const int key = nt * 16 + ncl;
        tv[nt] = TGT ? tmask[b * (cS * cT) + key * cT + t]
                     : tmask[b * (cSEED * cT) + key * cT + t];
    }

    for (int mt = 0; mt < 4; ++mt) {
        const int qrow = (TGT ? 64 : 0) + mt * 16;  // first query row of tile

        // ---- scores: A = q tile (16 x 32) ----------------------------------
        Frag a;
#pragma unroll
        for (int j = 0; j < 8; ++j) {
            const int k = aFragK(j, half);
            a.u32[j] = *(const unsigned int*)(qp + base + (size_t)(qrow + mrow) * cD + k);
        }

        v8f sc[NT];
#pragma unroll
        for (int nt = 0; nt < NT; ++nt) {
            v8f z = {};
            sc[nt] = __builtin_amdgcn_wmma_f32_16x16x32_bf16(false, a.v, false, kb[nt].v,
                                                             (short)0, z, false, false);
        }

        // ---- mask + scale, row max -----------------------------------------
        float m_r[8];
#pragma unroll
        for (int r = 0; r < 8; ++r) m_r[r] = -3.0e38f;
#pragma unroll
        for (int nt = 0; nt < NT; ++nt) {
            const int key = nt * 16 + ncl;
#pragma unroll
            for (int r = 0; r < 8; ++r) {
                int valid = tv[nt];
                if (TGT) {
                    const int q_idx = mt * 16 + r + half * 8;   // 0..63 target query
                    valid = valid && (mixmask[((b * cT + t) * 64 + q_idx) * 128 + key] != 0);
                }
                const float sv = valid ? sc[nt][r] * scale : -1.0e9f;
                sc[nt][r] = sv;
                m_r[r] = fmaxf(m_r[r], sv);
            }
        }
#pragma unroll
        for (int r = 0; r < 8; ++r)
#pragma unroll
            for (int msk = 1; msk <= 8; msk <<= 1)
                m_r[r] = fmaxf(m_r[r], __shfl_xor(m_r[r], msk, 32));

        // ---- exp + row sum + normalize + stage P to LDS --------------------
        float s_r[8];
#pragma unroll
        for (int r = 0; r < 8; ++r) s_r[r] = 0.0f;
#pragma unroll
        for (int nt = 0; nt < NT; ++nt)
#pragma unroll
            for (int r = 0; r < 8; ++r) {
                const float p = __expf(sc[nt][r] - m_r[r]);
                sc[nt][r] = p;
                s_r[r] += p;
            }
#pragma unroll
        for (int r = 0; r < 8; ++r) {
#pragma unroll
            for (int msk = 1; msk <= 8; msk <<= 1)
                s_r[r] += __shfl_xor(s_r[r], msk, 32);
            s_r[r] = 1.0f / s_r[r];
        }
#pragma unroll
        for (int nt = 0; nt < NT; ++nt)
#pragma unroll
            for (int r = 0; r < 8; ++r)
                p_lds[wid][r + half * 8][nt * 16 + ncl] = f2bf(sc[nt][r] * s_r[r]);

        __syncthreads();

        // ---- PV: A = P (from LDS), B = hoisted V, dual chains --------------
#pragma unroll
        for (int dn = 0; dn < 2; ++dn) {
            v8f o0 = {}, o1 = {};
#pragma unroll
            for (int kk = 0; kk < KK / 2; ++kk) {
                const int ka = kk, kbx = kk + KK / 2;
                Frag pa0, pa1;
#pragma unroll
                for (int j = 0; j < 8; ++j) {
                    const int kf = aFragK(j, half);
                    pa0.u32[j] = *(const unsigned int*)&p_lds[wid][mrow][ka * 32 + kf];
                    pa1.u32[j] = *(const unsigned int*)&p_lds[wid][mrow][kbx * 32 + kf];
                }
                o0 = __builtin_amdgcn_wmma_f32_16x16x32_bf16(false, pa0.v, false,
                                                             vbf[dn * KK + ka].v,
                                                             (short)0, o0, false, false);
                o1 = __builtin_amdgcn_wmma_f32_16x16x32_bf16(false, pa1.v, false,
                                                             vbf[dn * KK + kbx].v,
                                                             (short)0, o1, false, false);
            }
#pragma unroll
            for (int r = 0; r < 8; ++r) {
                const int srow = qrow + r + half * 8;   // global row in [0,S)
                xo[(size_t)((b * cS + srow) * cT + t) * cD + h * cDK + dn * 16 + ncl] =
                    f2bf(o0[r] + o1[r]);
            }
        }
        __syncthreads();   // protect p_lds before next m-tile
    }
}

// ---------------------------------------------------------------------------
// Kernel 4: out = Xattn @ Wd^T + bd, f32 output. Rows R over [B,S,T].
// ---------------------------------------------------------------------------
__global__ __launch_bounds__(256) void k_out(const unsigned short* __restrict__ xo,
                                             const unsigned short* __restrict__ wtd,
                                             const float* __restrict__ bd,
                                             float* __restrict__ out) {
    const int wid = threadIdx.x >> 5, lane = threadIdx.x & 31;
    const int mtile = blockIdx.x * 8 + wid;     // 0..4095
    const int half = lane >> 4, mrow = lane & 15, ncl = lane & 15;

    const unsigned short* xrow = xo + (size_t)(mtile * 16 + mrow) * cD;
    Frag a[8];
#pragma unroll
    for (int ks = 0; ks < 8; ++ks)
#pragma unroll
        for (int j = 0; j < 8; ++j) {
            const int k = ks * 32 + aFragK(j, half);
            a[ks].u32[j] = *(const unsigned int*)(xrow + k);
        }

    for (int nt = 0; nt < 16; ++nt) {
        v8f c0 = {}, c1 = {};
#pragma unroll
        for (int ks = 0; ks < 4; ++ks) {
            Frag bf0, bf1;
            const int k0 = ks * 32;
            const int k1 = (ks + 4) * 32;
#pragma unroll
            for (int j = 0; j < 8; ++j) {
                const int kf = aFragK(j, half);
                bf0.u16[2 * j]     = wtd[(k0 + kf) * 256 + nt * 16 + ncl];
                bf0.u16[2 * j + 1] = wtd[(k0 + kf + 1) * 256 + nt * 16 + ncl];
                bf1.u16[2 * j]     = wtd[(k1 + kf) * 256 + nt * 16 + ncl];
                bf1.u16[2 * j + 1] = wtd[(k1 + kf + 1) * 256 + nt * 16 + ncl];
            }
            c0 = __builtin_amdgcn_wmma_f32_16x16x32_bf16(false, a[ks].v, false, bf0.v,
                                                         (short)0, c0, false, false);
            c1 = __builtin_amdgcn_wmma_f32_16x16x32_bf16(false, a[ks + 4].v, false, bf1.v,
                                                         (short)0, c1, false, false);
        }
        const int ncol = nt * 16 + ncl;
        const float bias = bd[ncol];
#pragma unroll
        for (int r = 0; r < 8; ++r) {
            const int Ro = mtile * 16 + r + half * 8;
            out[(size_t)Ro * cD + ncol] = c0[r] + c1[r] + bias;
        }
    }
}

// ---------------------------------------------------------------------------
extern "C" void kernel_launch(void* const* d_in, const int* in_sizes, int n_in,
                              void* d_out, int out_size, void* d_ws, size_t ws_size,
                              hipStream_t stream) {
    (void)in_sizes; (void)n_in; (void)out_size; (void)ws_size;

    const float* query = (const float*)d_in[0];
    const float* key   = (const float*)d_in[1];
    const float* value = (const float*)d_in[2];
    // d_in[3] = seed_length (constant 64, compiled in)
    const int* src_mask        = (const int*)d_in[4];
    const int* src_target_mask = (const int*)d_in[5];
    const int* mix_mask        = (const int*)d_in[6];
    const float* Wq = (const float*)d_in[7];
    const float* Wk = (const float*)d_in[8];
    const float* Wv = (const float*)d_in[9];
    const float* Wd = (const float*)d_in[10];
    const float* bd = (const float*)d_in[11];
    float* out = (float*)d_out;

    // Workspace carve-up
    unsigned short* wt = (unsigned short*)d_ws;                 // 4 * 65536 bf16
    unsigned short* qp = (unsigned short*)((char*)d_ws + 524288);
    unsigned short* kT = qp + (size_t)16777216;
    unsigned short* vp = kT + (size_t)16777216;
    unsigned short* xo = vp + (size_t)16777216;

    // 1) weights -> bf16 transposed
    k_wt<<<1024, 256, 0, stream>>>(Wq, Wk, Wv, Wd, wt);

    // 2) QKV projections: 3 * 4096 M-tiles, 8 waves/block
    k_proj<<<1536, 256, 0, stream>>>(query, key, value, wt, qp, kT, vp);

    // 3) attention: 4096 (b,t,h) waves each, 4 waves/block
    k_attn<0><<<1024, 128, 0, stream>>>(qp, kT, vp, src_mask, nullptr, xo);
    k_attn<1><<<1024, 128, 0, stream>>>(qp, kT, vp, src_target_mask, mix_mask, xo);

    // 4) output projection + bias: 4096 M-tiles, 8 waves/block
    k_out<<<512, 256, 0, stream>>>(xo, wt + 3 * 65536, bd, out);
}
